// GAT_encoder_24438363914371
// MI455X (gfx1250) — matrix operand order repaired and checked by reading, not compiled, not run on hardware
//
#include <hip/hip_runtime.h>
#include <math.h>

// ---------- CDNA5 WMMA types ----------
typedef __attribute__((ext_vector_type(16))) __bf16 v16bf;
typedef __attribute__((ext_vector_type(8)))  float  v8f;

// f32 -> bf16 pair via native v_cvt_pk_bf16_f32
__device__ __forceinline__ unsigned int cvt2(float lo, float hi) {
  union { __bf16 b[2]; unsigned int u; } r;
  r.b[0] = (__bf16)lo; r.b[1] = (__bf16)hi;
  return r.u;
}

// ============================================================
// Kernel 0: pack f32 pairs -> bf16 (used once for W: 64 KB)
// ============================================================
__global__ void cvt_f32_to_bf16(const float* __restrict__ in,
                                unsigned int* __restrict__ out, int nPairs)
{
  int i = blockIdx.x * blockDim.x + threadIdx.x;
  if (i >= nPairs) return;
  float2 v = ((const float2*)in)[i];
  out[i] = cvt2(v.x, v.y);
}

// ============================================================
// Kernel 1: h[N,H] = x[N,F] @ W[H,F]^T  via v_wmma_f32_16x16x32_bf16,
// fused with a_s/a_d = h·att_src / h·att_dst.
//
// Phase 1 (producer): the block stages its 16-row A tile into LDS in
// PRE-SWIZZLED fragment order: uint4 pair #(k*32+lane) is exactly the
// 16-bf16 A fragment lane `lane` consumes at K-iteration k
//   (row = lane%16, K = {kA..kA+7, kA+16..kA+23}, kA = 32k + 8*(lane/16)).
// One conversion of the tile serves all 8 waves (8x less cvt VALU).
//
// Phase 2 (consumer): per K-step: 2x ds_load_b128 (A) + 2x global b128
// (bf16 W, pre-converted) + 1x WMMA. No VALU conversions in the loop.
//
// Epilogue: D stores + in-register attention logits:
//   a_s[row] += sum_col acc[row][col]*att_src[col]  (shfl_xor over half)
// ============================================================
__global__ void gat_gemm_wmma(const float* __restrict__ x,
                              const unsigned short* __restrict__ wbf,
                              const float* __restrict__ att_src,
                              const float* __restrict__ att_dst,
                              float* __restrict__ h,
                              float* __restrict__ a_s,
                              float* __restrict__ a_d,
                              int N, int F, int H)
{
  extern __shared__ unsigned int ldsA[];     // F*32 bytes (8 KB for F=256)

  const int tid  = threadIdx.x;
  const int lane = tid & 31;
  const int wave = tid >> 5;
  const int rowBase = blockIdx.x * 16;
  const int kIters  = F >> 5;                // F/32

  // ---- Phase 1: stage A tile (f32 -> bf16, fragment order) ----
  for (int fi = tid; fi < kIters * 32; fi += blockDim.x) {
    const int k    = fi >> 5;
    const int l    = fi & 31;
    const int row  = l & 15;
    const int kA   = k * 32 + (l >> 4) * 8;
    int grow = rowBase + row; if (grow >= N) grow = N - 1;   // clamp tail
    const float* src = x + (size_t)grow * F + kA;

    // speculative prefetch of the next block's matching fragment
    __builtin_prefetch(src + (size_t)16 * F, 0, 0);

    float4 f0 = *(const float4*)(src);        // K = kA..kA+3
    float4 f1 = *(const float4*)(src + 4);    // K = kA+4..kA+7
    float4 f2 = *(const float4*)(src + 16);   // K = kA+16..kA+19
    float4 f3 = *(const float4*)(src + 20);   // K = kA+20..kA+23

    uint4 u0 = { cvt2(f0.x, f0.y), cvt2(f0.z, f0.w), cvt2(f1.x, f1.y), cvt2(f1.z, f1.w) };
    uint4 u1 = { cvt2(f2.x, f2.y), cvt2(f2.z, f2.w), cvt2(f3.x, f3.y), cvt2(f3.z, f3.w) };

    unsigned int* dst = ldsA + (size_t)fi * 8;
    *(uint4*)(dst)     = u0;                  // ds_store_b128
    *(uint4*)(dst + 4) = u1;                  // ds_store_b128
  }
  __syncthreads();

  // ---- Phase 2: WMMA loop ----
  const int colBase = wave * 16;
  if (colBase >= H) return;                   // wave-uniform
  const int half = lane >> 4;
  const int l16  = lane & 15;
  const unsigned short* wr = wbf + (size_t)(colBase + l16) * F;

  v8f acc = {};
  #pragma unroll 2
  for (int k = 0; k < kIters; ++k) {
    union { v16bf v; uint4 q[2]; } A, B;
    const uint4* ap = (const uint4*)(ldsA + ((size_t)k * 32 + lane) * 8);
    A.q[0] = ap[0];                           // ds_load_b128
    A.q[1] = ap[1];                           // ds_load_b128

    const int kB = k * 32 + half * 8;
    B.q[0] = *(const uint4*)(wr + kB);        // 8 bf16: K = kB..kB+7
    B.q[1] = *(const uint4*)(wr + kB + 16);   // 8 bf16: K = kB+16..kB+23

    acc = __builtin_amdgcn_wmma_f32_16x16x32_bf16(
        /*neg_a=*/false, A.v, /*neg_b=*/false, B.v,
        /*c_mod=*/(short)0, acc, /*reuse_a=*/false, /*reuse_b=*/false);
  }

  // ---- Epilogue: D stores (tile-granular guard) ----
  const int col = colBase + l16;
  float* p = h + (size_t)(rowBase + half * 8) * H + col;
  if (rowBase + 16 <= N) {
    #pragma unroll
    for (int j = 0; j < 8; ++j) p[(size_t)j * H] = acc[j];
  } else {
    #pragma unroll
    for (int j = 0; j < 8; ++j)
      if (rowBase + half * 8 + j < N) p[(size_t)j * H] = acc[j];
  }

  // ---- Epilogue: fused attention logits ----
  const float asrc = att_src[col];
  const float adst = att_dst[col];
  #pragma unroll
  for (int j = 0; j < 8; ++j) {
    float vs = acc[j] * asrc;
    float vd = acc[j] * adst;
    #pragma unroll
    for (int off = 1; off < 16; off <<= 1) {  // reduce within 16-lane half
      vs += __shfl_xor(vs, off, 32);
      vd += __shfl_xor(vd, off, 32);
    }
    if (l16 == 0) {
      int row = rowBase + half * 8 + j;
      if (row < N) {
        atomicAdd(a_s + row, vs);
        atomicAdd(a_d + row, vd);
      }
    }
  }
}

// ============================================================
// Kernel 2: zero-init accumulators (every call -> deterministic)
// ============================================================
__global__ void init_bufs(float* __restrict__ out, float* __restrict__ m,
                          float* __restrict__ denom, float* __restrict__ a_s,
                          float* __restrict__ a_d, float* __restrict__ sums,
                          float* __restrict__ sumsq, long long totalNH, int N, int H)
{
  long long i = blockIdx.x * (long long)blockDim.x + threadIdx.x;
  if (i < totalNH) out[i] = 0.f;
  if (i < N) {
    m[i] = __uint_as_float(0xff800000u);  // -inf
    denom[i] = 0.f;
    a_s[i] = 0.f;
    a_d[i] = 0.f;
  }
  if (i < H) { sums[i] = 0.f; sumsq[i] = 0.f; }
}

// ============================================================
// Kernel 3: per-edge LeakyReLU(0.2) score + segment max over dst.
// Float atomic max via signed/unsigned int atomics (total order incl. -inf).
// Edges [0,E) from edge_index; [E, E+N) are self loops.
// ============================================================
__global__ void edge_score_max(const int* __restrict__ ei,
                               const float* __restrict__ a_s,
                               const float* __restrict__ a_d,
                               float* __restrict__ escore, float* __restrict__ m,
                               int E, int NE)
{
  int e = blockIdx.x * blockDim.x + threadIdx.x;
  if (e >= NE) return;
  int src = (e < E) ? ei[e]     : (e - E);
  int dst = (e < E) ? ei[E + e] : (e - E);
  float s = a_s[src] + a_d[dst];
  s = (s > 0.f) ? s : 0.2f * s;
  escore[e] = s;
  if (s >= 0.f) atomicMax((int*)(m + dst), __float_as_int(s));
  else          atomicMin((unsigned int*)(m + dst), __float_as_uint(s));
}

// ============================================================
// Kernel 4: ee = exp(e - m[dst]); denom[dst] += ee
// ============================================================
__global__ void edge_exp_sum(const int* __restrict__ ei,
                             const float* __restrict__ m,
                             float* __restrict__ escore,
                             float* __restrict__ denom, int E, int NE)
{
  int e = blockIdx.x * blockDim.x + threadIdx.x;
  if (e >= NE) return;
  int dst = (e < E) ? ei[E + e] : (e - E);
  float ee = __expf(escore[e] - m[dst]);
  escore[e] = ee;
  atomicAdd(denom + dst, ee);
}

// ============================================================
// Kernel 5: out[dst,:] += alpha * h[src,:]   (one wave per edge)
// h and out both fit in the 192 MB L2 -> gathers + atomics stay on-chip.
// ============================================================
__global__ void edge_aggregate(const int* __restrict__ ei,
                               const float* __restrict__ escore,
                               const float* __restrict__ denom,
                               const float* __restrict__ h,
                               float* __restrict__ out, int E, int NE, int H)
{
  int gid  = blockIdx.x * blockDim.x + threadIdx.x;
  int e    = gid >> 5;
  int lane = gid & 31;
  if (e >= NE) return;
  int src = (e < E) ? ei[e]     : (e - E);
  int dst = (e < E) ? ei[E + e] : (e - E);
  float alpha = escore[e] / (denom[dst] + 1e-16f);
  for (int f = lane * 4; f < H; f += 128) {
    float4 hv = *(const float4*)(h + (size_t)src * H + f);
    float* o = out + (size_t)dst * H + f;
    atomicAdd(o + 0, alpha * hv.x);
    atomicAdd(o + 1, alpha * hv.y);
    atomicAdd(o + 2, alpha * hv.z);
    atomicAdd(o + 3, alpha * hv.w);
  }
}

// ============================================================
// Kernel 6: y = relu(out + bias) in place; per-column sum / sumsq.
// Thread stride is a multiple of H -> each thread owns one column;
// one atomicAdd pair per thread at the end.
// ============================================================
__global__ void bias_relu_stats(float* __restrict__ y, const float* __restrict__ bias,
                                float* __restrict__ sums, float* __restrict__ sumsq,
                                long long total, int H)
{
  long long tid = blockIdx.x * (long long)blockDim.x + threadIdx.x;
  long long T   = (long long)gridDim.x * blockDim.x;
  int col0 = (int)(tid % H);
  if ((T % H) == 0) {
    float b = bias[col0];
    float ls = 0.f, lq = 0.f;
    for (long long i = tid; i < total; i += T) {
      float v = y[i] + b;
      v = (v > 0.f) ? v : 0.f;
      y[i] = v;
      ls += v; lq += v * v;
    }
    atomicAdd(sums + col0, ls);
    atomicAdd(sumsq + col0, lq);
  } else {
    for (long long i = tid; i < total; i += T) {
      int c = (int)(i % H);
      float v = y[i] + bias[c];
      v = (v > 0.f) ? v : 0.f;
      y[i] = v;
      atomicAdd(sums + c, v);
      atomicAdd(sumsq + c, v * v);
    }
  }
}

// ============================================================
// Kernel 7: y = (y - mu) * rsqrt(var + 1e-5) * gamma + beta
// One block per row (blockDim = H) -> no 64-bit modulo.
// ============================================================
__global__ void bn_normalize(float* __restrict__ y, const float* __restrict__ sums,
                             const float* __restrict__ sumsq,
                             const float* __restrict__ gamma,
                             const float* __restrict__ beta, int N, int H)
{
  int c   = threadIdx.x;
  int row = blockIdx.x;
  if (c >= H || row >= N) return;
  float invN = 1.0f / (float)N;
  float mu   = sums[c] * invN;
  float var  = sumsq[c] * invN - mu * mu;
  float inv  = rsqrtf(var + 1e-5f);
  size_t i = (size_t)row * H + c;
  y[i] = (y[i] - mu) * inv * gamma[c] + beta[c];
}

// ============================================================
// Launch
// ============================================================
extern "C" void kernel_launch(void* const* d_in, const int* in_sizes, int n_in,
                              void* d_out, int out_size, void* d_ws, size_t ws_size,
                              hipStream_t stream)
{
  const float* x       = (const float*)d_in[0];
  const float* W       = (const float*)d_in[1];
  const float* att_src = (const float*)d_in[2];
  const float* att_dst = (const float*)d_in[3];
  const float* bias    = (const float*)d_in[4];
  const float* gamma   = (const float*)d_in[5];
  const float* beta    = (const float*)d_in[6];
  const int*   ei      = (const int*)d_in[7];

  const int H  = in_sizes[2];            // 128
  const int F  = in_sizes[1] / H;        // 256
  const int N  = in_sizes[0] / F;        // 100000
  const int E  = in_sizes[7] / 2;        // 1600000
  const int NE = E + N;

  // Workspace layout (floats): ~60 MB
  float* ws     = (float*)d_ws;
  float* h      = ws;                         // N*H
  float* a_s    = h + (size_t)N * H;          // N
  float* a_d    = a_s + N;                    // N
  float* m      = a_d + N;                    // N
  float* denom  = m + N;                      // N
  float* escore = denom + N;                  // NE
  float* sums   = escore + NE;                // H
  float* sumsq  = sums + H;                   // H
  size_t wbfOff = (size_t)(sumsq + H - ws);
  wbfOff = (wbfOff + 3) & ~(size_t)3;         // 16B align for uint4 loads
  unsigned int* wbf = (unsigned int*)(ws + wbfOff);   // H*F/2 uints (64 KB)
  float* out = (float*)d_out;

  const int tb = 256;
  const long long totalNH = (long long)N * H;
  const int nhBlocks = (int)((totalNH + tb - 1) / tb);

  // 0. W -> bf16 (once per call; 64 KB)
  int wPairs = (H * F) / 2;
  cvt_f32_to_bf16<<<(wPairs + tb - 1) / tb, tb, 0, stream>>>(W, wbf, wPairs);

  // 1. zero accumulators (incl. a_s/a_d, which the GEMM accumulates into)
  init_bufs<<<nhBlocks, tb, 0, stream>>>(out, m, denom, a_s, a_d, sums, sumsq,
                                         totalNH, N, H);

  // 2. h = x @ W^T (bf16 WMMA, LDS-staged A) + fused attention logits
  const int colWaves = H / 16;               // 8 waves
  const int rowTiles = (N + 15) / 16;
  const int ldsBytes = F * 32;               // staged A tile (8 KB for F=256)
  gat_gemm_wmma<<<rowTiles, colWaves * 32, ldsBytes, stream>>>(
      x, (const unsigned short*)wbf, att_src, att_dst, h, a_s, a_d, N, F, H);

  // 3. edge scores + segment max
  edge_score_max<<<(NE + tb - 1) / tb, tb, 0, stream>>>(ei, a_s, a_d, escore, m, E, NE);

  // 4. exp + segment sum
  edge_exp_sum<<<(NE + tb - 1) / tb, tb, 0, stream>>>(ei, m, escore, denom, E, NE);

  // 5. scatter-aggregate (one wave per edge)
  long long aggThreads = (long long)NE * 32;
  edge_aggregate<<<(int)((aggThreads + tb - 1) / tb), tb, 0, stream>>>(
      ei, escore, denom, h, out, E, NE, H);

  // 6. bias + ReLU + batch stats (T = 131072, multiple of H=128)
  bias_relu_stats<<<512, tb, 0, stream>>>(out, bias, sums, sumsq, totalNH, H);

  // 7. batchnorm normalize (one block per row)
  bn_normalize<<<N, H, 0, stream>>>(out, sums, sumsq, gamma, beta, N, H);
}
